// DocRelPrompt_11699490914784
// MI455X (gfx1250) — matrix operand order repaired and checked by reading, not compiled, not run on hardware
//
#include <hip/hip_runtime.h>
#include <math.h>

#define B_ 16
#define L_ 2048
#define H_ 768
#define D_ 64
#define N_ 20
#define ROWS_TOT (2*N_ + L_)        // 2088 rows per batch in output
#define ROWS_PER_BLK 16
#define CHUNKS (L_/ROWS_PER_BLK)    // 128 partial-sum chunks per batch

typedef __attribute__((ext_vector_type(16))) _Float16 v16h;
typedef __attribute__((ext_vector_type(8)))  float    v8f;

// ---------------------------------------------------------------------------
// Kernel 1: embedding gather -> output copy, fused per-batch partial sums.
// Block = (chunk, b): copies 16 rows of 768 f32 with float4 (b128), prefetches
// the next random embedding row, and writes one float4 partial sum per thread.
// ---------------------------------------------------------------------------
__global__ __launch_bounds__(192) void k_gather(const int* __restrict__ ids,
                                                const float* __restrict__ wte,
                                                float* __restrict__ out,
                                                float* __restrict__ partials)
{
    const int t     = threadIdx.x;      // 0..191 -> columns [4t, 4t+4)
    const int chunk = blockIdx.x;       // 0..127
    const int b     = blockIdx.y;       // 0..15
    const int c0    = 4 * t;
    const int idbase = b * L_ + chunk * ROWS_PER_BLK;

    float4 acc = make_float4(0.f, 0.f, 0.f, 0.f);
    for (int r = 0; r < ROWS_PER_BLK; ++r) {
        const int tok = ids[idbase + r];
        if (r + 1 < ROWS_PER_BLK) {
            const int tok2 = ids[idbase + r + 1];
            __builtin_prefetch(wte + (size_t)tok2 * H_ + c0, 0, 3);
        }
        const float4 v = *(const float4*)(wte + (size_t)tok * H_ + c0);
        acc.x += v.x; acc.y += v.y; acc.z += v.z; acc.w += v.w;
        const size_t orow = (size_t)b * ROWS_TOT + 2 * N_ + chunk * ROWS_PER_BLK + r;
        *(float4*)(out + orow * H_ + c0) = v;
    }
    *(float4*)(partials + ((size_t)(b * CHUNKS + chunk)) * H_ + c0) = acc;
}

// ---------------------------------------------------------------------------
// Kernel 2: deterministic reduction of partials -> h_mean (B,H);
// also computes hs_rel = rel2 @ label_prompts (B,H).
// ---------------------------------------------------------------------------
__global__ __launch_bounds__(256) void k_reduce(const float* __restrict__ partials,
                                                const float* __restrict__ relevance,
                                                const float* __restrict__ label_prompts,
                                                float* __restrict__ hmean,
                                                float* __restrict__ hsrel)
{
    const int idx = blockIdx.x * 256 + threadIdx.x;   // 0 .. B*H-1
    const int b = idx / H_;
    const int h = idx % H_;
    float s = 0.f;
    for (int c = 0; c < CHUNKS; ++c)
        s += partials[((size_t)(b * CHUNKS + c)) * H_ + h];
    hmean[idx] = s * (1.0f / (float)L_);
    const float rel = relevance[b];
    hsrel[idx] = (1.f - rel) * label_prompts[h] + rel * label_prompts[H_ + h];
}

// ---------------------------------------------------------------------------
// WMMA 16x16 output tile of  C = A(rows base_m..)[MxK] * W^T  with K=H_=768,
// using v_wmma_f32_16x16x32_f16 (24 K-steps). A is row-major (M x 768) f32,
// W is row-major (64 x 768) f32 so B[k][n] = W[n*768 + k].
// Fragment layouts per CDNA5 ISA 7.12.2:
//   A (16-bit 16x32): k = ((e>>3)<<4) + (lane>>4)*8 + (e&7)
//      -> two contiguous 8-float runs: [k0 + 8*half, +8) and [k0+16+8*half, +8)
//   B (16-bit 32x16): k = ((e>>3)<<4) + (lane>>4)*8 + (e&7)
//      -> two contiguous 8-float runs: [k0 + 8*half, +8) and [k0+16+8*half, +8)
//   C (f32 16x16):    m = r + (lane>>4)*8, n = lane&15
// Rows beyond Mvalid are handled branch-free: clamp the address to row 0 and
// multiply the loaded values by 0 (keeps EXEC uniform, b128 loads).
// ---------------------------------------------------------------------------
__device__ inline v8f gemm_tile(const float* __restrict__ A, int Mvalid, int base_m,
                                const float* __restrict__ W, int base_n, int lane)
{
    v8f c = {};
    const int m16  = lane & 15;
    const int half = lane >> 4;
    const int m    = base_m + m16;
    const int n    = base_n + m16;
    const bool mv  = (m < Mvalid);
    const float amul = mv ? 1.0f : 0.0f;
    const float* arow = A + (size_t)(mv ? m : 0) * H_ + (half << 3);
    const float* brow = W + (size_t)n * H_ + (half << 3);

#pragma unroll 4
    for (int k0 = 0; k0 < H_; k0 += 32) {
        const float4 a0 = *(const float4*)(arow + k0);        // e = 0..3
        const float4 a1 = *(const float4*)(arow + k0 + 4);    // e = 4..7
        const float4 a2 = *(const float4*)(arow + k0 + 16);   // e = 8..11
        const float4 a3 = *(const float4*)(arow + k0 + 20);   // e = 12..15
        const float4 b0 = *(const float4*)(brow + k0);        // e = 0..3
        const float4 b1 = *(const float4*)(brow + k0 + 4);    // e = 4..7
        const float4 b2 = *(const float4*)(brow + k0 + 16);   // e = 8..11
        const float4 b3 = *(const float4*)(brow + k0 + 20);   // e = 12..15

        v16h a, bf;
        a[0]  = (_Float16)(a0.x * amul); a[1]  = (_Float16)(a0.y * amul);
        a[2]  = (_Float16)(a0.z * amul); a[3]  = (_Float16)(a0.w * amul);
        a[4]  = (_Float16)(a1.x * amul); a[5]  = (_Float16)(a1.y * amul);
        a[6]  = (_Float16)(a1.z * amul); a[7]  = (_Float16)(a1.w * amul);
        a[8]  = (_Float16)(a2.x * amul); a[9]  = (_Float16)(a2.y * amul);
        a[10] = (_Float16)(a2.z * amul); a[11] = (_Float16)(a2.w * amul);
        a[12] = (_Float16)(a3.x * amul); a[13] = (_Float16)(a3.y * amul);
        a[14] = (_Float16)(a3.z * amul); a[15] = (_Float16)(a3.w * amul);

        bf[0]  = (_Float16)b0.x; bf[1]  = (_Float16)b0.y;
        bf[2]  = (_Float16)b0.z; bf[3]  = (_Float16)b0.w;
        bf[4]  = (_Float16)b1.x; bf[5]  = (_Float16)b1.y;
        bf[6]  = (_Float16)b1.z; bf[7]  = (_Float16)b1.w;
        bf[8]  = (_Float16)b2.x; bf[9]  = (_Float16)b2.y;
        bf[10] = (_Float16)b2.z; bf[11] = (_Float16)b2.w;
        bf[12] = (_Float16)b3.x; bf[13] = (_Float16)b3.y;
        bf[14] = (_Float16)b3.z; bf[15] = (_Float16)b3.w;

        c = __builtin_amdgcn_wmma_f32_16x16x32_f16(false, a, false, bf,
                                                   (short)0, c, false, false);
    }
    return c;
}

// ---------------------------------------------------------------------------
// Kernel 3: one adapter. 8 waves:
//   Q (20x64, padded M=32): 2x4 tiles -> waves 0..7
//   K (16x64):              1x4 tiles -> waves 0..3
// Then scores = Q.K^T / 8, sigmoid -> probs (B*N) in workspace.
// ---------------------------------------------------------------------------
__global__ __launch_bounds__(256) void k_adapter(const float* __restrict__ prompts,
                                                 const float* __restrict__ qw,
                                                 const float* __restrict__ qb,
                                                 const float* __restrict__ kw,
                                                 const float* __restrict__ kb,
                                                 const float* __restrict__ ksrc,
                                                 float* __restrict__ probs_out)
{
    __shared__ float Qbuf[N_][D_];
    __shared__ float Kbuf[B_][D_];
    const int tid  = threadIdx.x;
    const int lane = tid & 31;
    const int wave = tid >> 5;

    {   // Q tiles: prompts (20 x 768) @ qw^T -> (20 x 64)
        const int bm = (wave >> 2) * 16;
        const int bn = (wave & 3) * 16;
        const v8f c = gemm_tile(prompts, N_, bm, qw, bn, lane);
        const int n = bn + (lane & 15);
        const float bias = qb[n];
        const int mo = (lane >> 4) * 8;
#pragma unroll
        for (int r = 0; r < 8; ++r) {
            const int m = bm + mo + r;
            if (m < N_) Qbuf[m][n] = c[r] + bias;
        }
    }
    if (wave < 4) {   // K tiles: ksrc (16 x 768) @ kw^T -> (16 x 64)
        const int bn = wave * 16;
        const v8f c = gemm_tile(ksrc, B_, 0, kw, bn, lane);
        const int n = bn + (lane & 15);
        const float bias = kb[n];
        const int mo = (lane >> 4) * 8;
#pragma unroll
        for (int r = 0; r < 8; ++r)
            Kbuf[mo + r][n] = c[r] + bias;
    }
    __syncthreads();

    for (int t = tid; t < B_ * N_; t += 256) {
        const int b = t / N_, n = t % N_;
        float s = 0.f;
#pragma unroll 8
        for (int d = 0; d < D_; ++d) s += Qbuf[n][d] * Kbuf[b][d];
        s *= 0.125f;                         // 1/sqrt(64)
        probs_out[t] = 1.0f / (1.0f + expf(-s));
    }
}

// ---------------------------------------------------------------------------
// Kernel 4: out[b, rowbase+n, :] = prompts[n,:] * probs[b,n] for both regions.
// ---------------------------------------------------------------------------
__global__ __launch_bounds__(256) void k_scale(const float* __restrict__ prompts,
                                               const float* __restrict__ probs_lbl,
                                               const float* __restrict__ probs_doc,
                                               float* __restrict__ out)
{
    const int idx = blockIdx.x * 256 + threadIdx.x;   // < 2*B*N*H
    const int a   = idx / (B_ * N_ * H_);
    const int rem = idx - a * (B_ * N_ * H_);
    const int b   = rem / (N_ * H_);
    const int r2  = rem - b * (N_ * H_);
    const int n   = r2 / H_;
    const int h   = r2 - n * H_;
    const float p = (a == 0) ? probs_lbl[b * N_ + n] : probs_doc[b * N_ + n];
    const int rowbase = (a == 0) ? 0 : N_;
    out[((size_t)b * ROWS_TOT + rowbase + n) * H_ + h] = prompts[n * H_ + h] * p;
}

// ---------------------------------------------------------------------------
extern "C" void kernel_launch(void* const* d_in, const int* in_sizes, int n_in,
                              void* d_out, int out_size, void* d_ws, size_t ws_size,
                              hipStream_t stream)
{
    const float* relevance     = (const float*)d_in[0];
    const int*   input_ids     = (const int*)  d_in[1];
    const float* wte           = (const float*)d_in[2];
    const float* prompts       = (const float*)d_in[3];
    const float* label_prompts = (const float*)d_in[4];
    const float* aq_w = (const float*)d_in[5];
    const float* aq_b = (const float*)d_in[6];
    const float* ak_w = (const float*)d_in[7];
    const float* ak_b = (const float*)d_in[8];
    const float* lq_w = (const float*)d_in[9];
    const float* lq_b = (const float*)d_in[10];
    const float* lk_w = (const float*)d_in[11];
    const float* lk_b = (const float*)d_in[12];

    float* out = (float*)d_out;
    float* ws  = (float*)d_ws;

    float* partials  = ws;                                   // B*CHUNKS*H f32
    float* hmean     = partials + (size_t)B_ * CHUNKS * H_;  // B*H
    float* hsrel     = hmean    + (size_t)B_ * H_;           // B*H
    float* probs_lbl = hsrel    + (size_t)B_ * H_;           // B*N
    float* probs_doc = probs_lbl + B_ * N_;                  // B*N

    // 1) gather/copy (bandwidth-bound ~200 MB) + fused partial sums
    k_gather<<<dim3(CHUNKS, B_), 192, 0, stream>>>(input_ids, wte, out, partials);
    // 2) deterministic mean + hs_rel
    k_reduce<<<(B_ * H_) / 256, 256, 0, stream>>>(partials, relevance, label_prompts,
                                                  hmean, hsrel);
    // 3) adapters (WMMA): label uses hs_rel, doc uses h_mean
    k_adapter<<<1, 256, 0, stream>>>(prompts, lq_w, lq_b, lk_w, lk_b, hsrel, probs_lbl);
    k_adapter<<<1, 256, 0, stream>>>(prompts, aq_w, aq_b, ak_w, ak_b, hmean, probs_doc);
    // 4) broadcast prompt scaling into output prompt regions
    k_scale<<<(2 * B_ * N_ * H_) / 256, 256, 0, stream>>>(prompts, probs_lbl,
                                                          probs_doc, out);
}
